// VRNN_71906342470242
// MI455X (gfx1250) — matrix operand order repaired and checked by reading, not compiled, not run on hardware
//
#include <hip/hip_runtime.h>
#include <cstdint>
#include <cstddef>

// ---------------------------------------------------------------------------
// VRNN forward, MI455X (gfx1250), wave32 + WMMA f32<-f16 16x16x32.
// One wave owns a 16-row batch tile and runs the entire T=128 recurrence with
// all state in LDS. Weights pre-packed once per launch into d_ws as f16 WMMA-B
// fragments. Rolled kt/stage loops + scheduling fences bound register
// pressure (<256 VGPRs, no spills) and keep static code I$-friendly.
// ---------------------------------------------------------------------------

typedef __attribute__((ext_vector_type(16))) _Float16 v16h;
typedef __attribute__((ext_vector_type(8)))  float    v8f;

#define T_STEPS 128
#define BATCH   2048
#define XD 2
#define ZD 10
#define HD 64
#define RD 64
#define LOG2PI 1.8378770664093453f

// d_in order: x, eps, then params leaves (nested dicts, keys sorted):
enum {
  IN_X = 0, IN_EPS,
  DEC_B1, DEC_B2, DEC_W1, DEC_W2,            // 'dec'        : b1,b2,w1,w2
  DECLV_B, DECLV_W,                          // 'dec_logvar' : b,w
  DECMU_B, DECMU_W,                          // 'dec_mean'   : b,w
  ENC_B1, ENC_B2, ENC_W1, ENC_W2,            // 'enc'
  ENCLV_B, ENCLV_W,                          // 'enc_logvar'
  ENCMU_B, ENCMU_W,                          // 'enc_mean'
  G0_BHH, G0_BIH, G0_WHH, G0_WIH,            // 'gru0'       : bhh,bih,whh,wih
  G1_BHH, G1_BIH, G1_WHH, G1_WIH,            // 'gru1'
  PHX_B1, PHX_B2, PHX_W1, PHX_W2,            // 'phi_x'
  PHZ_B1, PHZ_B2, PHZ_W1, PHZ_W2,            // 'phi_z'
  PRI_B1, PRI_B2, PRI_W1, PRI_W2,            // 'prior'
  PRILV_B, PRILV_W,                          // 'prior_logvar'
  PRIMU_B, PRIMU_W,                          // 'prior_mean'
  N_IN
};

struct Params { const float* p[N_IN]; };

// ---- packed-weight fragment bases (1 fragment = 512 f16 = 1KB, kt-major)
enum {
  F_PHX1 = 0,   F_PHX2 = 4,   F_PHZ1 = 12,  F_PHZ2 = 16,
  F_ENC1 = 24,  F_ENC2 = 40,  F_PRI1 = 48,  F_PRI2 = 56,
  F_DEC1 = 64,  F_DEC2 = 80,
  F_ENCMU = 88, F_ENCLV = 90, F_PRIMU = 92, F_PRILV = 94,
  F_DECMU = 96, F_DECLV = 98,
  F_G0IH = 100, F_G0HH = 148, F_G1IH = 172, F_G1HH = 196,
  F_TOTAL = 220                               // 220 KB in d_ws
};

enum { ACT_NONE = 0, ACT_SOFTPLUS = 1, ACT_SIGMOID = 2 };

__device__ __forceinline__ void sched_fence() { asm volatile("" ::: "memory"); }

__device__ __forceinline__ float sigmf(float x) { return 1.0f / (1.0f + __expf(-x)); }
__device__ __forceinline__ float softplusf(float x) {
  return x > 0.0f ? x + log1pf(__expf(-x)) : log1pf(__expf(x));
}

__device__ __forceinline__ v8f zero8() {
  v8f z = {0.f, 0.f, 0.f, 0.f, 0.f, 0.f, 0.f, 0.f};
  return z;
}

__device__ __forceinline__ void leaky8(v8f& c) {
#pragma unroll
  for (int v = 0; v < 8; ++v) c[v] = c[v] > 0.0f ? c[v] : 0.01f * c[v];
}

// ---- ISA A layout (16x32 f16): lane l row m=l&15; two 16B runs per lane.
__device__ __forceinline__ v16h load_A(const _Float16* A, int lda, int k0, int lane) {
  const int m  = lane & 15;
  const int kb = k0 + ((lane & 16) >> 1);
  const _Float16* p = A + m * lda + kb;
  union { uint4 q[2]; v16h v; } u;
  u.q[0] = *reinterpret_cast<const uint4*>(p);
  u.q[1] = *reinterpret_cast<const uint4*>(p + 16);
  return u.v;
}

// ---- pre-packed B fragment: lane l's 16 halves are contiguous 32B.
__device__ __forceinline__ v16h load_Bp(const _Float16* wsb, int frag, int lane) {
  const uint4* p =
      reinterpret_cast<const uint4*>(wsb + ((size_t)frag << 9) + (lane << 4));
  union { uint4 q[2]; v16h v; } u;
  u.q[0] = p[0];
  u.q[1] = p[1];
  return u.v;
}

// ---- C/D layout stores: lane l holds (m = v + (l>=16?8:0), n = nt*16+(l&15))
__device__ __forceinline__ void store_tile_h(_Float16* D, int ldd, int nt, const v8f& c, int lane) {
  const int n  = nt * 16 + (lane & 15);
  const int mb = (lane & 16) >> 1;
#pragma unroll
  for (int v = 0; v < 8; ++v) D[(mb + v) * ldd + n] = (_Float16)c[v];
}

__device__ __forceinline__ void store_tile_f(float* D, int ldd, int nt, const v8f& c, int lane) {
  const int n  = nt * 16 + (lane & 15);
  const int mb = (lane & 16) >> 1;
#pragma unroll
  for (int v = 0; v < 8; ++v) D[(mb + v) * ldd + n] = c[v];
}

// ---- generic [16 x 64] GEMM: A = logical concat [A0|A1] (each lda-wide),
//      runtime KT (1,2,4). Fragments kt-major with NT stride 4.
__device__ __forceinline__ void gemm4_dyn(const _Float16* A0, const _Float16* A1,
                                          int lda, int KT,
                                          const _Float16* wsb, int fragbase,
                                          const float* bias, int lane, v8f (&acc)[4]) {
#pragma unroll
  for (int nt = 0; nt < 4; ++nt) acc[nt] = zero8();
#pragma unroll 1
  for (int kt = 0; kt < KT; ++kt) {
    const _Float16* Ak = (kt < 2) ? A0 : A1;
    v16h a = load_A(Ak, lda, (kt & 1) * 32, lane);
#pragma unroll
    for (int nt = 0; nt < 4; ++nt) {
      v16h b = load_Bp(wsb, fragbase + kt * 4 + nt, lane);
      acc[nt] = __builtin_amdgcn_wmma_f32_16x16x32_f16(
          false, a, false, b, (short)0, acc[nt], false, false);
    }
  }
  const int n_l = lane & 15;
#pragma unroll
  for (int nt = 0; nt < 4; ++nt) {
    const float bv = bias[nt * 16 + n_l];
#pragma unroll
    for (int v = 0; v < 8; ++v) acc[nt][v] += bv;
  }
}

// ---- Linear->Leaky->Linear->Leaky; generic input; optional fp32 mirror.
__device__ __forceinline__ void mlp2_dyn(const _Float16* A0, const _Float16* A1,
                                         int lda1, int KT1,
                                         const _Float16* wsb, int fb1, const float* b1,
                                         int fb2, const float* b2,
                                         _Float16* hid, _Float16* outb,
                                         int lane, float* gout) {
  v8f acc[4];
  gemm4_dyn(A0, A1, lda1, KT1, wsb, fb1, b1, lane, acc);
#pragma unroll
  for (int nt = 0; nt < 4; ++nt) { leaky8(acc[nt]); store_tile_h(hid, 64, nt, acc[nt], lane); }
  sched_fence();
  gemm4_dyn(hid, hid, 64, 2, wsb, fb2, b2, lane, acc);
  const int n_l = lane & 15, mb = (lane & 16) >> 1;
#pragma unroll
  for (int nt = 0; nt < 4; ++nt) {
    leaky8(acc[nt]);
    store_tile_h(outb, 64, nt, acc[nt], lane);
    if (gout) {
      const int n = nt * 16 + n_l;
#pragma unroll
      for (int v = 0; v < 8; ++v) gout[(size_t)(mb + v) * HD + n] = acc[nt][v];
    }
  }
  sched_fence();
}

// ---- [16 x 192] gate GEMM in two NT=6 halves (bounded pressure);
//      A = logical concat [A0|A1], runtime KT (2 or 4), frag NT-stride 12.
__device__ __forceinline__ void gemm12_dyn(const _Float16* A0, const _Float16* A1,
                                           int KT, const _Float16* wsb, int fragbase,
                                           const float* bias, float* dst, int lane) {
#pragma unroll 1
  for (int half = 0; half < 2; ++half) {
    v8f acc[6];
#pragma unroll
    for (int nt = 0; nt < 6; ++nt) acc[nt] = zero8();
#pragma unroll 1
    for (int kt = 0; kt < KT; ++kt) {
      const _Float16* Ak = (kt < 2) ? A0 : A1;
      v16h a = load_A(Ak, 64, (kt & 1) * 32, lane);
#pragma unroll
      for (int nt = 0; nt < 6; ++nt) {
        v16h b = load_Bp(wsb, fragbase + kt * 12 + half * 6 + nt, lane);
        acc[nt] = __builtin_amdgcn_wmma_f32_16x16x32_f16(
            false, a, false, b, (short)0, acc[nt], false, false);
      }
    }
    const int n_l = lane & 15;
#pragma unroll
    for (int nt = 0; nt < 6; ++nt) {
      const int ng = half * 6 + nt;
      const float bv = bias[ng * 16 + n_l];
#pragma unroll
      for (int v = 0; v < 8; ++v) acc[nt][v] += bv;
      store_tile_f(dst, 192, ng, acc[nt], lane);
    }
    sched_fence();
  }
}

// ---- pair of small heads sharing one input (rolled): q=0 -> ACT0, q=1 -> ACT1
template <int NCOL, int ACT0, int ACT1>
__device__ __forceinline__ void head_pair(const _Float16* in, const _Float16* wsb,
                                          int fbA, const float* bA,
                                          int fbB, const float* bB,
                                          int lane, float* stA, float* stB,
                                          float* gpA, float* gpB) {
#pragma unroll 1
  for (int q = 0; q < 2; ++q) {
    const int fb       = q ? fbB : fbA;
    const float* bb    = q ? bB : bA;
    float* st          = q ? stB : stA;
    float* gp          = q ? gpB : gpA;
    v8f c = zero8();
#pragma unroll 1
    for (int kt = 0; kt < 2; ++kt) {
      v16h a = load_A(in, 64, kt * 32, lane);
      v16h b = load_Bp(wsb, fb + kt, lane);
      c = __builtin_amdgcn_wmma_f32_16x16x32_f16(
          false, a, false, b, (short)0, c, false, false);
    }
    const int n  = lane & 15;
    const int mb = (lane & 16) >> 1;
    if (n < NCOL) {
      const float bv = bb[n];
#pragma unroll
      for (int v = 0; v < 8; ++v) {
        float val = c[v] + bv;
        const int act = q ? ACT1 : ACT0;
        if (act == ACT_SOFTPLUS) val = softplusf(val);
        else if (act == ACT_SIGMOID) val = sigmf(val);
        st[(mb + v) * 16 + n] = val;
        if (gp) gp[(size_t)(mb + v) * NCOL + n] = val;
      }
    }
    sched_fence();
  }
}

// ---- GRU gate mixing (PyTorch r,z,n): gi/gh are [16 x 192] fp32 in LDS.
__device__ __forceinline__ void gru_elem(float* fh, _Float16* sh,
                                         const float* gi, const float* gh, int lane) {
#pragma unroll 4
  for (int i = 0; i < 32; ++i) {
    const int e = i * 32 + lane;
    const int m = e >> 6, f = e & 63;
    const float r  = sigmf(gi[m * 192 + f]       + gh[m * 192 + f]);
    const float zg = sigmf(gi[m * 192 + 64 + f]  + gh[m * 192 + 64 + f]);
    const float nn = tanhf(gi[m * 192 + 128 + f] + r * gh[m * 192 + 128 + f]);
    const float hn = (1.0f - zg) * nn + zg * fh[e];
    fh[e] = hn;
    sh[e] = (_Float16)hn;
  }
}

// ===========================================================================
// Weight pre-pack: fp32 row-major [out][in] -> f16 WMMA-B fragments in d_ws.
// ===========================================================================
struct PackDesc { const float* W; int indim; int outdim; int NT; int fragoff; };
struct PackAll  { PackDesc d[20]; };

__launch_bounds__(32)
__global__ void pack_weights(PackAll PA, _Float16* __restrict__ ws) {
  const int fid = blockIdx.x;
  int di = 0;
#pragma unroll
  for (int i = 1; i < 20; ++i)
    if (fid >= PA.d[i].fragoff) di = i;
  const PackDesc d = PA.d[di];
  const int local = fid - d.fragoff;
  const int kt = local / d.NT;
  const int nt = local - kt * d.NT;
  const int lane = threadIdx.x;
  const int n  = nt * 16 + (lane & 15);
  const int kb = kt * 32 + ((lane & 16) >> 1);
  union { uint4 q[2]; _Float16 h[16]; } u;
#pragma unroll
  for (int i = 0; i < 16; ++i) {
    const int k = kb + i + ((i & 8) ? 8 : 0);
    u.h[i] = (n < d.outdim && k < d.indim) ? (_Float16)d.W[(size_t)n * d.indim + k]
                                           : (_Float16)0.0f;
  }
  uint4* p = reinterpret_cast<uint4*>(ws + ((size_t)fid << 9) + (lane << 4));
  p[0] = u.q[0];
  p[1] = u.q[1];
}

// ===========================================================================
// Main persistent kernel
// ===========================================================================
__launch_bounds__(32)
__global__ void vrnn_persistent(Params P, const _Float16* __restrict__ wpk,
                                float* __restrict__ out) {
  const int lane = threadIdx.x;         // 0..31, one wave per block
  const int b0   = blockIdx.x * 16;     // 16-row batch tile

  const float* X   = P.p[IN_X];
  const float* EPS = P.p[IN_EPS];

  float* o_kld = out + 0;
  float* o_nll = out + 1;
  float* o_h   = out + 2;
  float* o_z   = o_h   + (size_t)2 * BATCH * RD;
  float* o_enc = o_z   + (size_t)BATCH * ZD;
  float* o_em  = o_enc + (size_t)BATCH * HD;
  float* o_elv = o_em  + (size_t)T_STEPS * BATCH * ZD;
  float* o_dm  = o_elv + (size_t)T_STEPS * BATCH * ZD;
  float* o_dlv = o_dm  + (size_t)T_STEPS * BATCH * XD;

  // ---- per-block (single wave) LDS: ~48 KB
  __shared__ __align__(16) _Float16 s_x   [16 * 32];   // x, K padded to 32
  __shared__ __align__(16) _Float16 s_z   [16 * 32];   // z, K padded to 32
  __shared__ __align__(16) _Float16 s_tmp [16 * 64];   // hidden / enc_t / prior_t / dec_t
  __shared__ __align__(16) _Float16 s_phix[16 * 64];
  __shared__ __align__(16) _Float16 s_phiz[16 * 64];
  __shared__ __align__(16) _Float16 s_h0  [16 * 64];   // f16 mirror of GRU state
  __shared__ __align__(16) _Float16 s_h1  [16 * 64];
  __shared__ float f_h0[16 * 64];                      // fp32 master GRU state
  __shared__ float f_h1[16 * 64];
  __shared__ float s_gi[16 * 192];
  __shared__ float s_gh[16 * 192];
  __shared__ float s_st0[16 * 16], s_st1[16 * 16], s_st2[16 * 16], s_st3[16 * 16];
  __shared__ float s_red[64];

  for (int i = lane; i < 1024; i += 32) {
    f_h0[i] = 0.0f; f_h1[i] = 0.0f;
    s_h0[i] = (_Float16)0.0f; s_h1[i] = (_Float16)0.0f;
  }
  for (int i = lane; i < 512; i += 32) { s_x[i] = (_Float16)0.0f; s_z[i] = (_Float16)0.0f; }

  float kacc = 0.0f, nacc = 0.0f;

#pragma unroll 1
  for (int t = 0; t < T_STEPS; ++t) {
    // prefetch next step's inputs (global_prefetch_b8)
    if (t + 1 < T_STEPS) {
      __builtin_prefetch(&X[((size_t)(t + 1) * BATCH + b0 + (lane & 15)) * XD], 0, 3);
      __builtin_prefetch(&EPS[((size_t)(t + 1) * BATCH + b0 + (lane & 15)) * ZD], 0, 3);
    }

    // ---- stage x_t (only live columns; padding zeroed once before loop)
    {
      const int m = lane >> 1, d = lane & 1;
      s_x[m * 32 + d] = (_Float16)X[((size_t)t * BATCH + b0 + m) * XD + d];
    }
    sched_fence();

    // ---- phi_x = MLP2(x)
    mlp2_dyn(s_x, s_x, 32, 1, wpk, F_PHX1, P.p[PHX_B1], F_PHX2, P.p[PHX_B2],
             s_tmp, s_phix, lane, nullptr);

    // ---- enc_t = MLP2([phi_x | h_top]); mirror fp32 encs_last at t=T-1
    mlp2_dyn(s_phix, s_h1, 64, 4, wpk, F_ENC1, P.p[ENC_B1], F_ENC2, P.p[ENC_B2],
             s_tmp, s_tmp, lane,
             (t == T_STEPS - 1) ? (o_enc + (size_t)b0 * HD) : nullptr);

    head_pair<ZD, ACT_NONE, ACT_SOFTPLUS>(
        s_tmp, wpk, F_ENCMU, P.p[ENCMU_B], F_ENCLV, P.p[ENCLV_B], lane,
        s_st0, s_st1,
        o_em + ((size_t)t * BATCH + b0) * ZD, o_elv + ((size_t)t * BATCH + b0) * ZD);

    // ---- prior from h_top
    mlp2_dyn(s_h1, s_h1, 64, 2, wpk, F_PRI1, P.p[PRI_B1], F_PRI2, P.p[PRI_B2],
             s_tmp, s_tmp, lane, nullptr);
    head_pair<ZD, ACT_NONE, ACT_SOFTPLUS>(
        s_tmp, wpk, F_PRIMU, P.p[PRIMU_B], F_PRILV, P.p[PRILV_B], lane,
        s_st2, s_st3, nullptr, nullptr);

    // ---- KLD partial: (lvp-lve) + exp(lve-lvp) + (me-mp)^2*exp(-lvp)
    for (int i = lane; i < 16 * ZD; i += 32) {
      const int m = i / ZD, d = i - m * ZD;
      const float me = s_st0[m * 16 + d], lve = s_st1[m * 16 + d];
      const float mp = s_st2[m * 16 + d], lvp = s_st3[m * 16 + d];
      const float dm = me - mp;
      kacc += (lvp - lve) + __expf(lve - lvp) + dm * dm * __expf(-lvp);
    }

    // ---- reparameterize z = eps*exp(0.5*lv) + mean
    for (int i = lane; i < 16 * ZD; i += 32) {
      const int m = i / ZD, d = i - m * ZD;
      const float zv = EPS[((size_t)t * BATCH + b0 + m) * ZD + d] *
                           __expf(0.5f * s_st1[m * 16 + d]) +
                       s_st0[m * 16 + d];
      s_z[m * 32 + d] = (_Float16)zv;
      if (t == T_STEPS - 1) o_z[(size_t)(b0 + m) * ZD + d] = zv;
    }
    sched_fence();

    // ---- phi_z = MLP2(z)
    mlp2_dyn(s_z, s_z, 32, 1, wpk, F_PHZ1, P.p[PHZ_B1], F_PHZ2, P.p[PHZ_B2],
             s_tmp, s_phiz, lane, nullptr);

    // ---- dec_t = MLP2([phi_z | h_top]); heads
    mlp2_dyn(s_phiz, s_h1, 64, 4, wpk, F_DEC1, P.p[DEC_B1], F_DEC2, P.p[DEC_B2],
             s_tmp, s_tmp, lane, nullptr);
    head_pair<XD, ACT_SIGMOID, ACT_SOFTPLUS>(
        s_tmp, wpk, F_DECMU, P.p[DECMU_B], F_DECLV, P.p[DECLV_B], lane,
        s_st2, s_st3,
        o_dm + ((size_t)t * BATCH + b0) * XD, o_dlv + ((size_t)t * BATCH + b0) * XD);

    // ---- NLL partial: (x-m)^2*exp(-(lv+0.001)) + lv (one element per lane)
    {
      const int m = lane >> 1, d = lane & 1;
      const float mu = s_st2[m * 16 + d], lv = s_st3[m * 16 + d];
      const float xv = X[((size_t)t * BATCH + b0 + m) * XD + d];
      const float df = xv - mu;
      nacc += df * df * __expf(-(lv + 0.001f)) + lv;
    }
    sched_fence();

    // ---- GRU stack (one rolled loop, both layers share the code path)
#pragma unroll 1
    for (int layer = 0; layer < 2; ++layer) {
      const _Float16* A0 = layer ? s_h0 : s_phix;   // gi input (concat for L0)
      const _Float16* A1 = layer ? s_h0 : s_phiz;
      const int kti      = layer ? 2 : 4;
      const int fb_i     = layer ? F_G1IH : F_G0IH;
      const int fb_h     = layer ? F_G1HH : F_G0HH;
      const float* bi    = layer ? P.p[G1_BIH] : P.p[G0_BIH];
      const float* bh    = layer ? P.p[G1_BHH] : P.p[G0_BHH];
      const _Float16* H  = layer ? s_h1 : s_h0;     // previous state (f16)
      float*    FH       = layer ? f_h1 : f_h0;     // fp32 master state
      _Float16* SH       = layer ? s_h1 : s_h0;

      gemm12_dyn(A0, A1, kti, wpk, fb_i, bi, s_gi, lane);
      gemm12_dyn(H, H, 2, wpk, fb_h, bh, s_gh, lane);
      gru_elem(FH, SH, s_gi, s_gh, lane);
      sched_fence();
    }
  }

  // ---- final hidden state h[2, B, 64]
  for (int i = lane; i < 1024; i += 32) {
    const int m = i >> 6, f = i & 63;
    o_h[((size_t)(b0 + m)) * RD + f]         = f_h0[i];
    o_h[((size_t)(BATCH + b0 + m)) * RD + f] = f_h1[i];
  }

  // ---- wave reduction of kld/nll partials, fold analytic constants, /B
  s_red[lane] = kacc;
  s_red[32 + lane] = nacc;
  if (lane == 0) {
    float sk = 0.0f, sn = 0.0f;
    for (int i = 0; i < 32; ++i) { sk += s_red[i]; sn += s_red[32 + i]; }
    const float rows_t = 16.0f * (float)T_STEPS;
    const float kldp = (0.5f * sk - 0.5f * (float)ZD * rows_t) / (float)BATCH;
    const float nllp = (0.5f * sn + 0.5f * (float)XD * LOG2PI * rows_t) / (float)BATCH;
    atomicAdd(o_kld, kldp);
    atomicAdd(o_nll, nllp);
  }
}

extern "C" void kernel_launch(void* const* d_in, const int* in_sizes, int n_in,
                              void* d_out, int out_size, void* d_ws, size_t ws_size,
                              hipStream_t stream) {
  (void)in_sizes; (void)n_in; (void)out_size; (void)ws_size; // needs >= 220 KB ws

  // kld/nll are atomically accumulated -> zero them each call (capture-safe).
  hipMemsetAsync(d_out, 0, 2 * sizeof(float), stream);

  Params P;
  for (int i = 0; i < N_IN; ++i) P.p[i] = (const float*)d_in[i];

  PackAll PA = {{
      {P.p[PHX_W1],   2, 64,  4, F_PHX1},
      {P.p[PHX_W2],  64, 64,  4, F_PHX2},
      {P.p[PHZ_W1],  10, 64,  4, F_PHZ1},
      {P.p[PHZ_W2],  64, 64,  4, F_PHZ2},
      {P.p[ENC_W1], 128, 64,  4, F_ENC1},
      {P.p[ENC_W2],  64, 64,  4, F_ENC2},
      {P.p[PRI_W1],  64, 64,  4, F_PRI1},
      {P.p[PRI_W2],  64, 64,  4, F_PRI2},
      {P.p[DEC_W1], 128, 64,  4, F_DEC1},
      {P.p[DEC_W2],  64, 64,  4, F_DEC2},
      {P.p[ENCMU_W], 64, 10,  1, F_ENCMU},
      {P.p[ENCLV_W], 64, 10,  1, F_ENCLV},
      {P.p[PRIMU_W], 64, 10,  1, F_PRIMU},
      {P.p[PRILV_W], 64, 10,  1, F_PRILV},
      {P.p[DECMU_W], 64,  2,  1, F_DECMU},
      {P.p[DECLV_W], 64,  2,  1, F_DECLV},
      {P.p[G0_WIH], 128, 192, 12, F_G0IH},
      {P.p[G0_WHH],  64, 192, 12, F_G0HH},
      {P.p[G1_WIH],  64, 192, 12, F_G1IH},
      {P.p[G1_WHH],  64, 192, 12, F_G1HH},
  }};

  pack_weights<<<dim3(F_TOTAL), dim3(32), 0, stream>>>(PA, (_Float16*)d_ws);
  vrnn_persistent<<<dim3(BATCH / 16), dim3(32), 0, stream>>>(
      P, (const _Float16*)d_ws, (float*)d_out);
}
